// VAE_5145370821224
// MI455X (gfx1250) — matrix operand order repaired and checked by reading, not compiled, not run on hardware
//
#include <hip/hip_runtime.h>
#include <cmath>

typedef __attribute__((ext_vector_type(16))) _Float16 v16h;
typedef __attribute__((ext_vector_type(8)))  _Float16 h8;
typedef __attribute__((ext_vector_type(8)))  float    v8f;

#define LA_C 1.0f
#define LG_C 0.5f

// ---------------- CDNA5 async global->LDS copy (16 bytes / lane) ----------------
__device__ __forceinline__ void async_copy_b128(unsigned lds_off, const void* gptr) {
  asm volatile("global_load_async_to_lds_b128 %0, %1, off"
               :: "v"(lds_off), "v"(gptr) : "memory");
}
__device__ __forceinline__ void async_wait0() {
  asm volatile("s_wait_asynccnt 0x0" ::: "memory");
}

// ---------------- WMMA helpers (wave32 layouts, ISA 7.12.2) ----------------
// A 16x32 f16 from row-major LDS [16][32]; per-lane elements are two contiguous
// 16B runs -> two h8 loads (ds_load_b128).
__device__ __forceinline__ v16h load_a16(const _Float16* lds) {
  int lane = threadIdx.x & 31;
  int M = lane & 15, hh = lane >> 4;
  h8 lo = *(const h8*)&lds[M * 32 + hh * 8];
  h8 hi = *(const h8*)&lds[M * 32 + 16 + hh * 8];
  return __builtin_shufflevector(lo, hi, 0, 1, 2, 3, 4, 5, 6, 7,
                                 8, 9, 10, 11, 12, 13, 14, 15);
}
__device__ __forceinline__ v8f wmma_f16(v16h a, v16h b, v8f c) {
  return __builtin_amdgcn_wmma_f32_16x16x32_f16(false, a, false, b, (short)0, c,
                                                false, false);
}
__device__ __forceinline__ int imin(int a, int b) { return a < b ? a : b; }

// ---------------- Generic WMMA GEMM ----------------
// C = [res +] act(A@W [+ bias]); optional scatter into padded QKV bufs.
// A [MxK], W [KxN] f32 row-major. Block tile 64(M) x 64(N), 4 waves; each wave
// owns one 16-col strip and 4 M-subtiles (4 accumulators sharing one B operand).
// A tile (64x32 f32, 8KB) staged via async DMA; B gathered directly from global.
__global__ __launch_bounds__(128) void gemm_wmma(
    const float* __restrict__ A, const float* __restrict__ W,
    const float* __restrict__ bias, const float* __restrict__ res,
    float* __restrict__ C, int M, int K, int N, int dogelu,
    float* __restrict__ Qs, float* __restrict__ Ks, float* __restrict__ Vs)
{
  __shared__ __align__(16) float Asf[64 * 32];
  const int tid = threadIdx.x;
  const int wv = tid >> 5, lane = tid & 31;
  const int m0 = blockIdx.x * 64;
  const int n0 = blockIdx.y * 64;
  const int wn0 = imin(n0 + wv * 16, N - 16);   // clamped wave column base
  const int Nc = lane & 15, hh = lane >> 4;
  v8f acc[4] = {v8f{}, v8f{}, v8f{}, v8f{}};
  for (int k0 = 0; k0 < K; k0 += 32) {
    // ---- A tile via async DMA: 512 16B granules, 4 per thread ----
#pragma unroll
    for (int gi = 0; gi < 4; ++gi) {
      int g = tid + gi * 128;
      int r = g >> 3, c4 = (g & 7) * 4;
      int gm = imin(m0 + r, M - 1);
      unsigned off = (unsigned)(uintptr_t)&Asf[r * 32 + c4];
      if (k0 + c4 < K) {
        async_copy_b128(off, &A[(size_t)gm * K + k0 + c4]);
      } else {
        *(float4*)&Asf[r * 32 + c4] = make_float4(0.f, 0.f, 0.f, 0.f);
      }
    }
    // ---- B operand: direct coalesced global gather (column walk), f32->f16 ----
    v16h b;
    if (k0 + hh * 16 + 15 < K) {
      const float* wp = &W[(size_t)(k0 + hh * 16) * N + wn0 + Nc];
#pragma unroll
      for (int i = 0; i < 16; ++i) b[i] = (_Float16)wp[(size_t)i * N];
    } else {
#pragma unroll
      for (int i = 0; i < 16; ++i) {
        int kk = k0 + hh * 16 + i;
        b[i] = (_Float16)((kk < K) ? W[(size_t)kk * N + wn0 + Nc] : 0.f);
      }
    }
    if (k0 + 32 < K)  // speculative prefetch of next A slab
      __builtin_prefetch(&A[(size_t)imin(m0 + (tid & 63), M - 1) * K + k0 + 32], 0, 0);
    async_wait0();
    __syncthreads();
    // ---- 4 M-subtiles per wave, one WMMA each ----
#pragma unroll
    for (int mt = 0; mt < 4; ++mt) {
      const float* ap = &Asf[(mt * 16 + Nc) * 32 + hh * 16];
      float4 f0 = *(const float4*)ap;
      float4 f1 = *(const float4*)(ap + 4);
      float4 f2 = *(const float4*)(ap + 8);
      float4 f3 = *(const float4*)(ap + 12);
      v16h a;
      a[0]=(_Float16)f0.x;  a[1]=(_Float16)f0.y;  a[2]=(_Float16)f0.z;  a[3]=(_Float16)f0.w;
      a[4]=(_Float16)f1.x;  a[5]=(_Float16)f1.y;  a[6]=(_Float16)f1.z;  a[7]=(_Float16)f1.w;
      a[8]=(_Float16)f2.x;  a[9]=(_Float16)f2.y;  a[10]=(_Float16)f2.z; a[11]=(_Float16)f2.w;
      a[12]=(_Float16)f3.x; a[13]=(_Float16)f3.y; a[14]=(_Float16)f3.z; a[15]=(_Float16)f3.w;
      acc[mt] = wmma_f16(a, b, acc[mt]);
    }
    __syncthreads();
  }
  // ---- epilogue ----
  const int gn = n0 + wv * 16 + Nc;
#pragma unroll
  for (int mt = 0; mt < 4; ++mt) {
#pragma unroll
    for (int r = 0; r < 8; ++r) {
      int gm = m0 + mt * 16 + r + 8 * hh;
      if (gm < M && gn < N) {
        float v = acc[mt][r];
        if (bias) v += bias[gn];
        if (dogelu) v = 0.5f * v * (1.f + erff(v * 0.70710678118654752f));
        if (res) v += res[(size_t)gm * N + gn];
        if (Qs) {  // scatter column gn = h*39 + j*13 + d into padded [bh][n][16]
          int hq = gn / 39, rem = gn - hq * 39;
          int j = rem / 13, d = rem - j * 13;
          int bb = gm >> 10, n = gm & 1023;
          float* dst = (j == 0) ? Qs : (j == 1) ? Ks : Vs;
          dst[(((size_t)(bb * 16 + hq)) * 1024 + n) * 16 + d] = v;
        } else {
          C[(size_t)gm * N + gn] = v;
        }
      }
    }
  }
}

// ---------------- LayerNorm (last dim 512), one block per row ----------------
__global__ __launch_bounds__(256) void ln_kernel(
    const float* __restrict__ x, const float* __restrict__ g,
    const float* __restrict__ b, float* __restrict__ y)
{
  __shared__ float red[256];
  const int row = blockIdx.x;
  const float* xr = x + (size_t)row * 512;
  float v0 = xr[threadIdx.x], v1 = xr[threadIdx.x + 256];
  red[threadIdx.x] = v0 + v1;
  __syncthreads();
  for (int o = 128; o > 0; o >>= 1) {
    if (threadIdx.x < o) red[threadIdx.x] += red[threadIdx.x + o];
    __syncthreads();
  }
  float mean = red[0] * (1.f / 512.f);
  __syncthreads();
  float d0 = v0 - mean, d1 = v1 - mean;
  red[threadIdx.x] = d0 * d0 + d1 * d1;
  __syncthreads();
  for (int o = 128; o > 0; o >>= 1) {
    if (threadIdx.x < o) red[threadIdx.x] += red[threadIdx.x + o];
    __syncthreads();
  }
  float rstd = rsqrtf(red[0] * (1.f / 512.f) + 1e-5f);
  float* yr = y + (size_t)row * 512;
  yr[threadIdx.x]       = d0 * rstd * g[threadIdx.x]       + b[threadIdx.x];
  yr[threadIdx.x + 256] = d1 * rstd * g[threadIdx.x + 256] + b[threadIdx.x + 256];
}

// ---------------- Mean pool: [B,1024,512] -> [B,512] ----------------
__global__ __launch_bounds__(512) void pool_kernel(
    const float* __restrict__ y, float* __restrict__ out)
{
  int b = blockIdx.x, d = threadIdx.x;
  float s = 0.f;
  for (int n = 0; n < 1024; ++n) s += y[((size_t)b * 1024 + n) * 512 + d];
  out[b * 512 + d] = s * (1.f / 1024.f);
}

// ---------------- small utility kernels ----------------
__global__ void fill_zero4(float4* p, int n4) {
  int i = blockIdx.x * blockDim.x + threadIdx.x;
  if (i < n4) p[i] = make_float4(0.f, 0.f, 0.f, 0.f);
}
__global__ void make_fmask(const int* __restrict__ mask, float* __restrict__ fm, int n) {
  int i = blockIdx.x * blockDim.x + threadIdx.x;
  if (i < n) fm[i] = mask[i] ? 1.f : 0.f;
}

// ---------------- Fused attention ----------------
// Padded zeroed bufs Q/K/V: [bh=64][n=1024][16] f32. One wave per (b,h,16-q tile).
// Two passes: (1) row max of masked dots + (adj*colmask)@V, (2) exp/sum + P@V.
__global__ __launch_bounds__(32) void attn_kernel(
    const float* __restrict__ Qb, const float* __restrict__ Kb,
    const float* __restrict__ Vb, const float* __restrict__ fmask,
    const float* __restrict__ adj, float* __restrict__ outp)
{
  __shared__ __align__(16) _Float16 Pt[16 * 32];
  const int blk = blockIdx.x;
  const int bh = blk >> 6;
  const int b  = bh >> 4;
  const int h  = bh & 15;
  const int q0 = (blk & 63) << 4;
  const int lane = threadIdx.x;
  const int Nc = lane & 15, hh = lane >> 4;
  const float scale = 0.27735009811261457f;  // 13^-0.5 folded into Q

  // Q A-operand: rows q0+Nc, dims hh*8..hh*8+7 (pad dims zero); upper K half zero.
  v16h aq;
  {
    const float4* qp =
        (const float4*)&Qb[(((size_t)bh * 1024) + q0 + Nc) * 16 + hh * 8];
    float4 qa = qp[0], qb2 = qp[1];
    float qv[8] = {qa.x, qa.y, qa.z, qa.w, qb2.x, qb2.y, qb2.z, qb2.w};
#pragma unroll
    for (int i = 0; i < 8; ++i) {
      aq[i] = (_Float16)(qv[i] * scale);
      aq[i + 8] = (_Float16)0.f;
    }
  }
  unsigned rmask = 0;  // query-row mask bits for rows r+8*hh
#pragma unroll
  for (int r = 0; r < 8; ++r)
    if (fmask[b * 1024 + q0 + r + 8 * hh] != 0.f) rmask |= (1u << r);

  auto load_bk = [&](int kt) -> v16h {  // K^T B-operand, 16 keys at kt
    float kv[16] = {};
    {
      const float4* kp = (const float4*)&Kb[(((size_t)bh * 1024) + kt + Nc) * 16 + hh * 8];
      float4 a0 = kp[0], a1 = kp[1];
      kv[0]=a0.x; kv[1]=a0.y; kv[2]=a0.z; kv[3]=a0.w;
      kv[4]=a1.x; kv[5]=a1.y; kv[6]=a1.z; kv[7]=a1.w;
    }
    v16h bk;
#pragma unroll
    for (int i = 0; i < 16; ++i) bk[i] = (_Float16)kv[i];
    return bk;
  };
  auto load_bv = [&](int k0) -> v16h {  // V B-operand, 32 keys at k0
    v16h bv;
#pragma unroll
    for (int i = 0; i < 16; ++i)
      bv[i] = (_Float16)Vb[(((size_t)bh * 1024) + k0 + hh * 8 + (i < 8 ? i : 8 + i)) * 16 + Nc];
    return bv;
  };
  auto load_adj = [&](int k0) -> v16h {  // (adj * colmask) A-operand, 32 keys
    const size_t rb = ((size_t)(b * 1024 + q0 + Nc)) * 1024 + k0 + hh * 8;
    const int fb = b * 1024 + k0 + hh * 8;
    const float4* ap = (const float4*)&adj[rb];
    const float4* aph = (const float4*)&adj[rb + 16];
    const float4* fp = (const float4*)&fmask[fb];
    const float4* fph = (const float4*)&fmask[fb + 16];
    float4 a0 = ap[0], a1 = ap[1], a2 = aph[0], a3 = aph[1];
    float4 f0 = fp[0], f1 = fp[1], f2 = fph[0], f3 = fph[1];
    v16h aa;
    aa[0]=(_Float16)(a0.x*f0.x); aa[1]=(_Float16)(a0.y*f0.y);
    aa[2]=(_Float16)(a0.z*f0.z); aa[3]=(_Float16)(a0.w*f0.w);
    aa[4]=(_Float16)(a1.x*f1.x); aa[5]=(_Float16)(a1.y*f1.y);
    aa[6]=(_Float16)(a1.z*f1.z); aa[7]=(_Float16)(a1.w*f1.w);
    aa[8]=(_Float16)(a2.x*f2.x); aa[9]=(_Float16)(a2.y*f2.y);
    aa[10]=(_Float16)(a2.z*f2.z); aa[11]=(_Float16)(a2.w*f2.w);
    aa[12]=(_Float16)(a3.x*f3.x); aa[13]=(_Float16)(a3.y*f3.y);
    aa[14]=(_Float16)(a3.z*f3.z); aa[15]=(_Float16)(a3.w*f3.w);
    return aa;
  };

  float mrow[8], lrow[8];
  v8f acc_adj = {};
#pragma unroll
  for (int r = 0; r < 8; ++r) { mrow[r] = -3.0e38f; lrow[r] = 0.f; }

  // ---- pass 1: row max + adjacency term ----
  for (int k0 = 0; k0 < 1024; k0 += 32) {
    v16h aa = load_adj(k0);
    v16h bv = load_bv(k0);
    acc_adj = wmma_f16(aa, bv, acc_adj);
#pragma unroll
    for (int t = 0; t < 2; ++t) {
      int kt = k0 + t * 16;
      v16h bk = load_bk(kt);
      v8f s = {};
      s = wmma_f16(aq, bk, s);
      bool cm = fmask[b * 1024 + kt + Nc] != 0.f;
#pragma unroll
      for (int r = 0; r < 8; ++r) {
        float sv = (((rmask >> r) & 1u) && cm) ? s[r] : -3.0e38f;
        float mx = sv;
        mx = fmaxf(mx, __shfl_xor(mx, 1, 32));
        mx = fmaxf(mx, __shfl_xor(mx, 2, 32));
        mx = fmaxf(mx, __shfl_xor(mx, 4, 32));
        mx = fmaxf(mx, __shfl_xor(mx, 8, 32));
        mrow[r] = fmaxf(mrow[r], mx);
      }
    }
  }

  // ---- pass 2: P = exp(S-m), row sums, P@V via LDS relayout ----
  v8f acc_p = {};
  for (int k0 = 0; k0 < 1024; k0 += 32) {
    v16h bv = load_bv(k0);
#pragma unroll
    for (int t = 0; t < 2; ++t) {
      int kt = k0 + t * 16;
      v16h bk = load_bk(kt);
      v8f s = {};
      s = wmma_f16(aq, bk, s);
      bool cm = fmask[b * 1024 + kt + Nc] != 0.f;
#pragma unroll
      for (int r = 0; r < 8; ++r) {
        float sv = (((rmask >> r) & 1u) && cm) ? s[r] : -3.0e38f;
        float p = __expf(sv - mrow[r]);
        float ss = p;
        ss += __shfl_xor(ss, 1, 32);
        ss += __shfl_xor(ss, 2, 32);
        ss += __shfl_xor(ss, 4, 32);
        ss += __shfl_xor(ss, 8, 32);
        lrow[r] += ss;
        Pt[(r + 8 * hh) * 32 + t * 16 + Nc] = (_Float16)p;  // C-layout -> A tile
      }
    }
    __syncthreads();
    v16h ap = load_a16(Pt);
    acc_p = wmma_f16(ap, bv, acc_p);
    __syncthreads();
  }

#pragma unroll
  for (int r = 0; r < 8; ++r) {
    if (Nc < 13) {
      int gm = q0 + r + 8 * hh;
      float at = ((rmask >> r) & 1u) ? acc_adj[r] : 0.f;
      float o = LA_C * acc_p[r] / lrow[r] + LG_C * at;
      outp[((size_t)(b * 1024 + gm)) * 208 + h * 13 + Nc] = o;
    }
  }
}

// ---------------- Driver ----------------
extern "C" void kernel_launch(void* const* d_in, const int* in_sizes, int n_in,
                              void* d_out, int out_size, void* d_ws, size_t ws_size,
                              hipStream_t stream)
{
  (void)in_sizes; (void)n_in; (void)out_size; (void)ws_size;
  const float* x       = (const float*)d_in[0];
  const int*   mask    = (const int*)  d_in[1];
  const float* adj     = (const float*)d_in[2];
  const float* embed_w = (const float*)d_in[3];
  const float* embed_b = (const float*)d_in[4];
  const float* ln1_g   = (const float*)d_in[5];
  const float* ln1_b   = (const float*)d_in[6];
  const float* qkv_w   = (const float*)d_in[7];
  const float* out_w   = (const float*)d_in[8];
  const float* out_b   = (const float*)d_in[9];
  const float* ln2_g   = (const float*)d_in[10];
  const float* ln2_b   = (const float*)d_in[11];
  const float* ff1_w   = (const float*)d_in[12];
  const float* ff1_b   = (const float*)d_in[13];
  const float* ff2_w   = (const float*)d_in[14];
  const float* ff2_b   = (const float*)d_in[15];
  const float* lnout_g = (const float*)d_in[16];
  const float* lnout_b = (const float*)d_in[17];
  const float* ffo1_w  = (const float*)d_in[18];
  const float* ffo1_b  = (const float*)d_in[19];
  const float* ffo2_w  = (const float*)d_in[20];
  const float* ffo2_b  = (const float*)d_in[21];
  float* out = (float*)d_out;

  float* ws  = (float*)d_ws;
  float* h   = ws;  ws += (size_t)4096 * 512;      // residual stream
  float* y   = ws;  ws += (size_t)4096 * 512;      // LN output
  float* Qb  = ws;  ws += (size_t)64 * 1024 * 16;  // padded Q [bh][n][16]
  float* Kb  = ws;  ws += (size_t)64 * 1024 * 16;  // padded K
  float* Vb  = ws;  ws += (size_t)64 * 1024 * 16;  // padded V
  float* ao  = ws;  ws += (size_t)4096 * 208;      // attention out (b,n,h,d)
  float* fm  = ws;  ws += (size_t)4096 * 2048;     // FF hidden
  float* pl  = ws;  ws += (size_t)4 * 512;         // pooled
  float* f2  = ws;  ws += (size_t)4 * 2048;        // head FF hidden
  float* fmk = ws;                                 // fmask [4096]

  dim3 blk128(128);
  fill_zero4<<<3072, 256, 0, stream>>>((float4*)Qb, 3 * 64 * 1024 * 4);
  make_fmask<<<16, 256, 0, stream>>>(mask, fmk, 4096);

  // h = x @ embed_w + embed_b
  gemm_wmma<<<dim3(64, 8), blk128, 0, stream>>>(
      x, embed_w, embed_b, nullptr, h, 4096, 512, 512, 0, nullptr, nullptr, nullptr);
  for (int l = 0; l < 2; ++l) {
    ln_kernel<<<4096, 256, 0, stream>>>(h, ln1_g + l * 512, ln1_b + l * 512, y);
    gemm_wmma<<<dim3(64, 10), blk128, 0, stream>>>(   // qkv with padded scatter
        y, qkv_w + (size_t)l * 512 * 624, nullptr, nullptr, nullptr,
        4096, 512, 624, 0, Qb, Kb, Vb);
    attn_kernel<<<4096, 32, 0, stream>>>(Qb, Kb, Vb, fmk, adj, ao);
    gemm_wmma<<<dim3(64, 8), blk128, 0, stream>>>(    // h += ao @ out_w + out_b
        ao, out_w + (size_t)l * 208 * 512, out_b + l * 512, h, h,
        4096, 208, 512, 0, nullptr, nullptr, nullptr);
    ln_kernel<<<4096, 256, 0, stream>>>(h, ln2_g + l * 512, ln2_b + l * 512, y);
    gemm_wmma<<<dim3(64, 32), blk128, 0, stream>>>(
        y, ff1_w + (size_t)l * 512 * 2048, ff1_b + l * 2048, nullptr, fm,
        4096, 512, 2048, 1, nullptr, nullptr, nullptr);
    gemm_wmma<<<dim3(64, 8), blk128, 0, stream>>>(
        fm, ff2_w + (size_t)l * 2048 * 512, ff2_b + l * 512, h, h,
        4096, 2048, 512, 0, nullptr, nullptr, nullptr);
  }
  ln_kernel<<<4096, 256, 0, stream>>>(h, lnout_g, lnout_b, y);
  pool_kernel<<<4, 512, 0, stream>>>(y, pl);
  gemm_wmma<<<dim3(1, 32), blk128, 0, stream>>>(
      pl, ffo1_w, ffo1_b, nullptr, f2, 4, 512, 2048, 1, nullptr, nullptr, nullptr);
  gemm_wmma<<<dim3(1, 8), blk128, 0, stream>>>(
      f2, ffo2_w, ffo2_b, nullptr, out, 4, 2048, 512, 0, nullptr, nullptr, nullptr);
}
